// lstm_probing_40037685133522
// MI455X (gfx1250) — matrix operand order repaired and checked by reading, not compiled, not run on hardware
//
#include <hip/hip_runtime.h>
#include <math.h>

// ---------------------------------------------------------------------------
// LSTM decoder for MI455X (gfx1250, wave32).
// Recurrence: fp32 WMMA 16x16x4 (precision compounds over 128 steps).
// Output projection: bf16 WMMA 16x16x32, fp32 accumulate, TDM-staged LDS.
// B=32, T=128, H=1024, V=32000, G=4H=4096.
// ---------------------------------------------------------------------------

typedef __attribute__((ext_vector_type(2)))  float    v2f;
typedef __attribute__((ext_vector_type(8)))  float    v8f;
typedef __attribute__((ext_vector_type(8)))  __bf16   v8bf;
typedef __attribute__((ext_vector_type(16))) __bf16   v16bf;
typedef __attribute__((ext_vector_type(4)))  unsigned u32x4;
typedef __attribute__((ext_vector_type(4)))  int      i32x4;
typedef __attribute__((ext_vector_type(8)))  int      i32x8;

#define HH 1024
#define BB 32
#define TT 128
#define VV 32000
#define GG 4096

__device__ __forceinline__ v8f wmma4(v2f a, v2f b, v8f c) {
  return __builtin_amdgcn_wmma_f32_16x16x4_f32(false, a, false, b, (short)0, c,
                                               false, false);
}

__device__ __forceinline__ v8f wmma32bf(v16bf a, v16bf b, v8f c) {
  return __builtin_amdgcn_wmma_f32_16x16x32_bf16(false, a, false, b, (short)0,
                                                 c, false, false);
}

__device__ __forceinline__ v16bf cat8(v8bf a, v8bf b) {
  return __builtin_shufflevector(a, b, 0, 1, 2, 3, 4, 5, 6, 7, 8, 9, 10, 11,
                                 12, 13, 14, 15);
}

__device__ __forceinline__ v8f vzero8() {
  v8f z = {0.f, 0.f, 0.f, 0.f, 0.f, 0.f, 0.f, 0.f};
  return z;
}

__device__ __forceinline__ float sigmoidf_(float x) {
  return 1.0f / (1.0f + __expf(-x));
}

// ---------------------------------------------------------------------------
// TDM: issue a 2D tensor tile load (bf16 elements) into LDS with padding:
// 16 DWORDs (64B) copied, then 4 DWORDs (16B) of pad -> 80B LDS row pitch.
// Dims/strides in data_size (2B) units.  Caller is one full wave; completion
// tracked with TENSORcnt.
// ---------------------------------------------------------------------------
__device__ __forceinline__ void tdm_load_2d(unsigned lds_off, const void* gptr,
                                            unsigned tensor_d0,
                                            unsigned tensor_d1,
                                            unsigned tile_d0, unsigned tile_d1,
                                            unsigned stride_d0) {
  unsigned long long ga = (unsigned long long)gptr;
  u32x4 g0;
  g0.x = 1u;                                                // count=1 (valid)
  g0.y = lds_off;                                           // lds_addr (bytes)
  g0.z = (unsigned)ga;                                      // global_addr lo
  g0.w = (unsigned)((ga >> 32) & 0x1FFFFFFu) | (2u << 30);  // addr hi | type=2

  i32x8 g1;
  // data_size=1 (2B) | pad_enable | pad_interval=3 (64B) | pad_amount=3 (16B)
  g1[0] = (int)((1u << 16) | (1u << 20) | (3u << 22) | (3u << 25));
  g1[1] = (int)(tensor_d0 << 16);                           // dim0 lo
  g1[2] = (int)((tensor_d0 >> 16) | (tensor_d1 << 16));     // dim0 hi|dim1 lo
  g1[3] = (int)((tensor_d1 >> 16) | (tile_d0 << 16));       // dim1 hi|tile0
  g1[4] = (int)(tile_d1 & 0xFFFFu);                         // tile1, tile2=0
  g1[5] = (int)stride_d0;                                   // dim0 stride lo
  g1[6] = 0;                                                // stride hi, s1 lo
  g1[7] = 0;

  i32x4 g2;  // tensor_dim2=1, tensor_dim3=1, stride2=0, tile_dim3=0
  g2[0] = 1; g2[1] = 1; g2[2] = 0; g2[3] = 0;
  i32x4 g3;  // tensor_dim3_stride=0, tensor_dim4=1, tile_dim4=0
  g3[0] = 0; g3[1] = 1 << 16; g3[2] = 0; g3[3] = 0;

#if defined(__clang_major__) && __clang_major__ >= 23
  i32x8 g4 = {0, 0, 0, 0, 0, 0, 0, 0};
  __builtin_amdgcn_tensor_load_to_lds(g0, g1, g2, g3, g4, 0);
#else
  __builtin_amdgcn_tensor_load_to_lds(g0, g1, g2, g3, 0);
#endif
}

// ---------------------------------------------------------------------------
// Kernel 1: fold W_ih[:,1:] + W_hh -> Wc, extract x-column wx, combine biases,
//           reset grid-barrier state (every launch; ws is not re-poisoned).
// ---------------------------------------------------------------------------
__global__ __launch_bounds__(256) void k_prep(const float* __restrict__ W_ih,
                                              const float* __restrict__ W_hh,
                                              const float* __restrict__ b_ih,
                                              const float* __restrict__ b_hh,
                                              float* __restrict__ Wc,
                                              float* __restrict__ wx,
                                              float* __restrict__ bc,
                                              unsigned* __restrict__ bar) {
  unsigned i = blockIdx.x * 256u + threadIdx.x;  // 0 .. G*H-1
  unsigned j = i >> 10;
  unsigned k = i & 1023u;
  Wc[i] = W_ih[(size_t)j * 1025u + 1u + k] + W_hh[i];
  if (k == 0) {
    wx[j] = W_ih[(size_t)j * 1025u];
    bc[j] = b_ih[j] + b_hh[j];
  }
  if (i < 2) bar[i] = 0u;
}

// ---------------------------------------------------------------------------
// Kernel 2: W_out fp32 -> bf16 (RNE).
// ---------------------------------------------------------------------------
__global__ __launch_bounds__(256) void k_wconv(const float* __restrict__ src,
                                               __bf16* __restrict__ dst) {
  size_t i = ((size_t)blockIdx.x * 256u + threadIdx.x) * 8u;
#pragma unroll
  for (int u = 0; u < 8; ++u) dst[i + u] = (__bf16)src[i + u];
}

// ---------------------------------------------------------------------------
// Kernel 3: h0 = enc @ W_h^T, c0 = enc @ W_c^T via fp32 WMMA.  256 waves.
// ---------------------------------------------------------------------------
__global__ __launch_bounds__(256) void k_inithc(const float* __restrict__ enc,
                                                const float* __restrict__ W_h,
                                                const float* __restrict__ W_c,
                                                float* __restrict__ h0,
                                                float* __restrict__ c0) {
  unsigned wid  = blockIdx.x * 8u + (threadIdx.x >> 5);  // 0..255
  unsigned lane = threadIdx.x & 31u;
  unsigned col  = lane & 15u;
  unsigned hf   = lane >> 4;
  unsigned koff = hf * 2u;

  unsigned sel = wid >> 7;
  unsigned tw  = wid & 127u;
  unsigned M0  = (tw >> 6) * 16u;
  unsigned N0  = (tw & 63u) * 16u;

  const float* W   = sel ? W_c : W_h;
  float*       dst = sel ? c0 : h0;

  const float* Ab = enc + (size_t)(M0 + col) * HH + koff;
  const float* Bb = W + (size_t)(N0 + col) * HH + koff;

  v8f acc = vzero8();
#pragma unroll 4
  for (int kk = 0; kk < HH / 4; ++kk) {
    v2f a = *(const v2f*)(Ab + kk * 4);
    v2f b = *(const v2f*)(Bb + kk * 4);
    acc = wmma4(a, b, acc);
  }
#pragma unroll
  for (int r = 0; r < 8; ++r)
    dst[(size_t)(M0 + r + hf * 8u) * HH + N0 + col] = acc[r];
}

// ---------------------------------------------------------------------------
// Kernel 4: persistent LSTM recurrence.  16 WGs x 8 waves = 128 waves, each
// owning one 16x16 tile of (h, c); c lives in 8 VGPRs for all 128 steps.
// One device-scope barrier per step; h double-buffered; hs stored as bf16.
// ---------------------------------------------------------------------------
__global__ __launch_bounds__(256) void k_lstm(const float* __restrict__ y,
                                              const float* __restrict__ Wc,
                                              const float* __restrict__ wx,
                                              const float* __restrict__ bc,
                                              float* __restrict__ hb0,
                                              float* __restrict__ hb1,
                                              const float* __restrict__ c0,
                                              __bf16* __restrict__ hsbf,
                                              unsigned* __restrict__ bar) {
  const unsigned NWG = 16u;
  unsigned wid  = blockIdx.x * 8u + (threadIdx.x >> 5);  // 0..127
  unsigned lane = threadIdx.x & 31u;
  unsigned col  = lane & 15u;
  unsigned hf   = lane >> 4;
  unsigned koff = hf * 2u;
  unsigned M0   = (wid >> 6) * 16u;   // batch tile
  unsigned N0   = (wid & 63u) * 16u;  // hidden-column tile

  v8f creg;
#pragma unroll
  for (int r = 0; r < 8; ++r)
    creg[r] = c0[(size_t)(M0 + r + hf * 8u) * HH + N0 + col];

  for (int t = 0; t < TT; ++t) {
    const float* hin  = (t & 1) ? hb1 : hb0;
    float*       hout = (t & 1) ? hb0 : hb1;
    const float* Ab   = hin + (size_t)(M0 + col) * HH + koff;

    v8f acc[4];
#pragma unroll
    for (int q = 0; q < 4; ++q) {
      v8f a8 = vzero8();
      const float* Bb = Wc + (size_t)(q * HH + N0 + col) * HH + koff;
#pragma unroll 4
      for (int kk = 0; kk < HH / 4; ++kk) {
        v2f a = *(const v2f*)(Ab + kk * 4);
        v2f b = *(const v2f*)(Bb + kk * 4);
        a8 = wmma4(a, b, a8);
      }
      acc[q] = a8;
    }

    float w0 = wx[0 * HH + N0 + col], w1 = wx[1 * HH + N0 + col];
    float w2 = wx[2 * HH + N0 + col], w3 = wx[3 * HH + N0 + col];
    float bi = bc[0 * HH + N0 + col], bf = bc[1 * HH + N0 + col];
    float bg = bc[2 * HH + N0 + col], bo_ = bc[3 * HH + N0 + col];

#pragma unroll
    for (int r = 0; r < 8; ++r) {
      unsigned row = M0 + r + hf * 8u;  // batch index 0..31
      float x  = y[(size_t)row * TT + t];
      float ig = sigmoidf_(acc[0][r] + x * w0 + bi);
      float fg = sigmoidf_(acc[1][r] + x * w1 + bf);
      float gg = tanhf(acc[2][r] + x * w2 + bg);
      float og = sigmoidf_(acc[3][r] + x * w3 + bo_);
      float cn = fg * creg[r] + ig * gg;
      creg[r]  = cn;
      float hn = og * tanhf(cn);
      hout[(size_t)row * HH + N0 + col] = hn;
      hsbf[((size_t)t * BB + row) * HH + N0 + col] = (__bf16)hn;
    }

    // device-scope sense-reversing barrier
    __syncthreads();
    if (threadIdx.x == 0) {
      __threadfence();
      unsigned g = __hip_atomic_load(&bar[1], __ATOMIC_RELAXED,
                                     __HIP_MEMORY_SCOPE_AGENT);
      unsigned a = __hip_atomic_fetch_add(&bar[0], 1u, __ATOMIC_ACQ_REL,
                                          __HIP_MEMORY_SCOPE_AGENT);
      if (a == NWG - 1u) {
        __hip_atomic_store(&bar[0], 0u, __ATOMIC_RELAXED,
                           __HIP_MEMORY_SCOPE_AGENT);
        __hip_atomic_fetch_add(&bar[1], 1u, __ATOMIC_RELEASE,
                               __HIP_MEMORY_SCOPE_AGENT);
      } else {
        while (__hip_atomic_load(&bar[1], __ATOMIC_ACQUIRE,
                                 __HIP_MEMORY_SCOPE_AGENT) == g)
          __builtin_amdgcn_s_sleep(2);
      }
    }
    __syncthreads();
  }
}

// ---------------------------------------------------------------------------
// Kernel 5: logits = hs @ W_out^T + b_out as [B, V, T], bf16 WMMA 16x16x32.
// WG tile 64(M) x 256(N); rows remapped to b*T + t (constant b per WG since
// Mg is 64-aligned) so A-panels are 2D TDM tiles and every 64B output line is
// fully written.  Double-buffered LDS panels filled by the Tensor Data Mover
// (wave 0 issues, TENSORcnt-tracked); 40-elem pitch from TDM pad config gives
// conflict-free ds_load_b128 fragment reads.  8 waves in 2x4 grid, each
// computing 32x64 (2 A-frags x 4 B-frags -> 8 WMMAs / K-tile).
// ---------------------------------------------------------------------------
#define KT 32  // K-depth per tile
#define LP 40  // LDS row pitch (elems) = 64B data + 16B TDM pad

__global__ __launch_bounds__(256) void k_out_bf(const __bf16* __restrict__ hsbf,
                                                const __bf16* __restrict__ Wobf,
                                                const float* __restrict__ bo,
                                                float* __restrict__ out) {
  __shared__ __bf16 As[2][64][LP];   // 10.0 KB
  __shared__ __bf16 Bs[2][256][LP];  // 40.0 KB

  unsigned tid  = threadIdx.x;
  unsigned gm   = blockIdx.x / 125u, gn = blockIdx.x % 125u;
  unsigned Mg   = gm * 64u;           // global row base (row = b*T + t)
  unsigned Vg   = gn * 256u;          // global vocab base
  unsigned warp = tid >> 5, lane = tid & 31u;
  unsigned col  = lane & 15u, hf = lane >> 4;
  unsigned Moff = (warp >> 2) * 32u;  // wave's M offset in WG tile
  unsigned Noff = (warp & 3u) * 64u;  // wave's N offset in WG tile

  // A tile: constant b, contiguous t (Mg is 64-aligned, T=128)
  unsigned bblk = Mg >> 7, tb = Mg & 127u;
  const __bf16* Abase = hsbf + (((size_t)tb * BB + bblk) << 10);
  const __bf16* Bbase = Wobf + ((size_t)Vg << 10);

  v8f acc[2][4];
#pragma unroll
  for (int i = 0; i < 2; ++i)
#pragma unroll
    for (int j = 0; j < 4; ++j) acc[i][j] = vzero8();

  if (warp == 0) {
    tdm_load_2d((unsigned)(size_t)&As[0][0][0], Abase, HH, (unsigned)TT * BB,
                KT, 64u, (unsigned)BB * HH);
    tdm_load_2d((unsigned)(size_t)&Bs[0][0][0], Bbase, HH, VV, KT, 256u, HH);
    __builtin_amdgcn_s_wait_tensorcnt(0);
  }
  __syncthreads();

  for (int kt = 0; kt < HH / KT; ++kt) {
    unsigned bi = (unsigned)kt & 1u;
    bool more = (kt + 1) < (HH / KT);
    if (warp == 0 && more) {
      unsigned K0 = (unsigned)(kt + 1) * KT;
      tdm_load_2d((unsigned)(size_t)&As[bi ^ 1u][0][0], Abase + K0, HH,
                  (unsigned)TT * BB, KT, 64u, (unsigned)BB * HH);
      tdm_load_2d((unsigned)(size_t)&Bs[bi ^ 1u][0][0], Bbase + K0, HH, VV, KT,
                  256u, HH);
    }

    // A fragments: per-lane chunks at K = hf*8 and hf*8+16 (ISA 16-bit A map)
    v16bf af[2];
#pragma unroll
    for (int i = 0; i < 2; ++i) {
      unsigned ar = Moff + (unsigned)i * 16u + col;
      v8bf lo = *(const v8bf*)&As[bi][ar][hf * 8u];
      v8bf hi = *(const v8bf*)&As[bi][ar][hf * 8u + 16u];
      af[i] = cat8(lo, hi);
    }
    // B fragments: per-lane chunks at K = hf*8 and hf*8+16 (matches A)
#pragma unroll
    for (int j = 0; j < 4; ++j) {
      unsigned br = Noff + (unsigned)j * 16u + col;
      v8bf lo = *(const v8bf*)&Bs[bi][br][hf * 8u];
      v8bf hi = *(const v8bf*)&Bs[bi][br][hf * 8u + 16u];
      v16bf bf = cat8(lo, hi);
      acc[0][j] = wmma32bf(af[0], bf, acc[0][j]);
      acc[1][j] = wmma32bf(af[1], bf, acc[1][j]);
    }

    if (warp == 0 && more) __builtin_amdgcn_s_wait_tensorcnt(0);
    __syncthreads();
  }

#pragma unroll
  for (int j = 0; j < 4; ++j) {
    unsigned v  = Vg + Noff + (unsigned)j * 16u + col;
    float    bv = bo[v];
#pragma unroll
    for (int i = 0; i < 2; ++i) {
#pragma unroll
      for (int r = 0; r < 8; ++r) {
        unsigned R = Mg + Moff + (unsigned)i * 16u + r + hf * 8u;
        unsigned b = R >> 7, t = R & 127u;
        out[((size_t)b * VV + v) * TT + t] = acc[i][j][r] + bv;
      }
    }
  }
}

// ---------------------------------------------------------------------------
extern "C" void kernel_launch(void* const* d_in, const int* in_sizes, int n_in,
                              void* d_out, int out_size, void* d_ws,
                              size_t ws_size, hipStream_t stream) {
  const float* y    = (const float*)d_in[0];  // [32,128]
  const float* enc  = (const float*)d_in[1];  // [32,1024]
  const float* W_h  = (const float*)d_in[2];  // [1024,1024]
  const float* W_c  = (const float*)d_in[3];  // [1024,1024]
  const float* W_ih = (const float*)d_in[4];  // [4096,1025]
  const float* W_hh = (const float*)d_in[5];  // [4096,1024]
  const float* b_ih = (const float*)d_in[6];  // [4096]
  const float* b_hh = (const float*)d_in[7];  // [4096]
  const float* W_o  = (const float*)d_in[8];  // [32000,1024]
  const float* b_o  = (const float*)d_in[9];  // [32000]
  float* out = (float*)d_out;                 // [32,32000,128]
  float* ws  = (float*)d_ws;

  // workspace layout: ~91 MB
  unsigned* bar = (unsigned*)ws;                   // 2 (pad to 4 floats)
  float* Wc  = ws + 4;                             // 4096*1024
  float* wx  = Wc + (size_t)GG * HH;               // 4096
  float* bc  = wx + GG;                            // 4096
  float* hb0 = bc + GG;                            // 32*1024
  float* hb1 = hb0 + (size_t)BB * HH;              // 32*1024
  float* c0  = hb1 + (size_t)BB * HH;              // 32*1024
  __bf16* hsbf = (__bf16*)(c0 + (size_t)BB * HH);  // 128*32*1024 bf16
  __bf16* Wobf = hsbf + (size_t)TT * BB * HH;      // 32000*1024 bf16

  k_prep<<<dim3((GG * HH) / 256), dim3(256), 0, stream>>>(
      W_ih, W_hh, b_ih, b_hh, Wc, wx, bc, bar);
  k_wconv<<<dim3(((size_t)VV * HH) / (256 * 8)), dim3(256), 0, stream>>>(W_o,
                                                                         Wobf);
  k_inithc<<<dim3(32), dim3(256), 0, stream>>>(enc, W_h, W_c, hb0, c0);
  k_lstm<<<dim3(16), dim3(256), 0, stream>>>(y, Wc, wx, bc, hb0, hb1, c0, hsbf,
                                             bar);
  // 64 M-blocks x 125 N-blocks
  k_out_bf<<<dim3(8000), dim3(256), 0, stream>>>(hsbf, Wobf, b_o, out);
}
